// GraphEnhancedDetect_72636486910244
// MI455X (gfx1250) — compile-verified, matrix operand order checked by reading, not run on hardware
//
#include <hip/hip_runtime.h>

typedef __attribute__((ext_vector_type(16))) _Float16 v16h;
typedef __attribute__((ext_vector_type(8)))  _Float16 v8h;
typedef __attribute__((ext_vector_type(8)))  float    v8f;
typedef __attribute__((ext_vector_type(4)))  unsigned int v4u;
typedef __attribute__((ext_vector_type(8)))  int v8i;
typedef __attribute__((ext_vector_type(4)))  int v4i;

#define DEV __device__ __forceinline__

// CDNA5 16-bit fragment K mapping (ISA 7.12.2): lane<16 -> K in {0..7, 16..23},
// lane>=16 -> K in {8..15, 24..31}; half j=0..7 maps to first run, j=8..15 to second.
DEV int kfrag(int lane, int j) { return ((lane & 16) ? 8 : 0) + ((j < 8) ? j : (j + 8)); }

DEV v16h cat8(v8h lo, v8h hi) {
    return __builtin_shufflevector(lo, hi, 0,1,2,3,4,5,6,7,8,9,10,11,12,13,14,15);
}

// ---------------------------------------------------------------------------
// Prep: pack conv weights (BN1 scale folded) into WMMA B-fragment order:
// wp[(((tap*8+chunk)*16+ntile)*32+lane)*16 + j]  with B[k][n] = w[cout][cin][ky][kx]*s
// ---------------------------------------------------------------------------
__global__ void pack_conv_kernel(const float* __restrict__ conv_w, const float* __restrict__ conv_b,
                                 const float* __restrict__ g, const float* __restrict__ be,
                                 const float* __restrict__ m, const float* __restrict__ v,
                                 _Float16* __restrict__ wp, float* __restrict__ bias2) {
    int idx = blockIdx.x * 256 + threadIdx.x;
    if (idx < 9 * 8 * 16 * 32 * 16) {
        int j    = idx & 15;
        int lane = (idx >> 4) & 31;
        int nt   = (idx >> 9) & 15;
        int ch   = (idx >> 13) & 7;
        int tap  = idx >> 16;          // 0..8
        int ky = tap / 3, kx = tap % 3;
        int cout = nt * 16 + (lane & 15);
        int cin  = ch * 32 + kfrag(lane, j);
        float s = g[cout] * rsqrtf(v[cout] + 1e-5f);
        wp[idx] = (_Float16)(conv_w[((cout * 256 + cin) * 3 + ky) * 3 + kx] * s);
    }
    if (idx < 256) {
        float s = g[idx] * rsqrtf(v[idx] + 1e-5f);
        bias2[idx] = conv_b[idx] * s + be[idx] - m[idx] * s;
    }
}

// Pack wq/wk/wv/wo (out = in @ W.T  =>  B[k][n] = W[n][k]) into fragment order.
__global__ void pack_lin_kernel(const float* __restrict__ wq, const float* __restrict__ wk,
                                const float* __restrict__ wv, const float* __restrict__ wo,
                                _Float16* __restrict__ lp) {
    int idx = blockIdx.x * 256 + threadIdx.x;
    if (idx >= 4 * 8 * 16 * 32 * 16) return;
    int j    = idx & 15;
    int lane = (idx >> 4) & 31;
    int nt   = (idx >> 9) & 15;
    int ch   = (idx >> 13) & 7;
    int mat  = idx >> 16;              // 0..3
    int n = nt * 16 + (lane & 15);
    int k = ch * 32 + kfrag(lane, j);
    const float* W = (mat == 0) ? wq : (mat == 1) ? wk : (mat == 2) ? wv : wo;
    lp[idx] = (_Float16)W[n * 256 + k];
}

// ---------------------------------------------------------------------------
// Fused conv3x3 + BN1 + SiLU, implicit GEMM with WMMA, double-buffered LDS
// pipeline. One block per (b, y) row: M=80 pixels (5 tiles), N=256 (16 tiles).
// 8 waves x (5 M-tiles x 2 N-tiles). 24 pipeline stages = 3 tap-rows x 8 chunks.
// feat written channel-last f16: feat[((b*80+y)*80+x)*256 + c].
// ---------------------------------------------------------------------------
__launch_bounds__(256)
__global__ void conv_wmma_kernel(const float* __restrict__ x, const _Float16* __restrict__ wp,
                                 const float* __restrict__ bias2, _Float16* __restrict__ feat) {
    __shared__ _Float16 sh[2][82 * 32];   // [buf][col 0..81][cin 0..31]
    int b = blockIdx.x / 80, y = blockIdx.x % 80;
    int tid = threadIdx.x, lane = tid & 31, wave = tid >> 5;

    v8f acc[5][2];
#pragma unroll
    for (int mt = 0; mt < 5; ++mt)
#pragma unroll
        for (int jj = 0; jj < 2; ++jj) acc[mt][jj] = {};

    int base = (lane & 16) ? 8 : 0;
    float stash[11];

    // stage s (0..23): ky = s>>3, chunk = s&7
    auto stage_load = [&](int s) {
        int ky = s >> 3, ch = s & 7;
        int ys = y + ky - 1;
        bool rowok = (ys >= 0) && (ys < 80);
#pragma unroll
        for (int t = 0; t < 11; ++t) {
            int i = tid + t * 256;
            float val = 0.f;
            if (i < 82 * 32) {
                int ci = i / 82, col = i % 82, xg = col - 1;
                if (rowok && xg >= 0 && xg < 80)
                    val = x[((b * 256 + ch * 32 + ci) * 80 + ys) * 80 + xg];
            }
            stash[t] = val;
        }
    };
    auto stage_store = [&](int s) {
        _Float16* dst = sh[s & 1];
#pragma unroll
        for (int t = 0; t < 11; ++t) {
            int i = tid + t * 256;
            if (i < 82 * 32) {
                int ci = i / 82, col = i % 82;
                dst[col * 32 + ci] = (_Float16)stash[t];
            }
        }
    };

    stage_load(0);
    stage_store(0);
    __syncthreads();

    for (int s = 0; s < 24; ++s) {
        if (s + 1 < 24) stage_load(s + 1);        // hide latency under WMMAs
        int ky = s >> 3, ch = s & 7;
        const _Float16* cur = sh[s & 1];
#pragma unroll
        for (int kx = 0; kx < 3; ++kx) {
            int tap = ky * 3 + kx;
            v16h bf[2];
#pragma unroll
            for (int jj = 0; jj < 2; ++jj) {
                int nt = wave * 2 + jj;
                bf[jj] = *(const v16h*)(wp + ((((tap * 8 + ch) * 16 + nt) * 32 + lane) << 4));
            }
#pragma unroll
            for (int mt = 0; mt < 5; ++mt) {
                int col = mt * 16 + (lane & 15) + kx;
                const _Float16* p = cur + col * 32 + base;
                v16h af = cat8(*(const v8h*)p, *(const v8h*)(p + 16));
#pragma unroll
                for (int jj = 0; jj < 2; ++jj)
                    acc[mt][jj] = __builtin_amdgcn_wmma_f32_16x16x32_f16(
                        false, af, false, bf[jj], (short)0, acc[mt][jj], false, false);
            }
        }
        if (s + 1 < 24) stage_store(s + 1);
        __syncthreads();
    }

    // Epilogue: +bias, SiLU, f16 store channel-last.
    int mrow = (lane & 16) ? 8 : 0;
#pragma unroll
    for (int mt = 0; mt < 5; ++mt)
#pragma unroll
        for (int jj = 0; jj < 2; ++jj) {
            int cout = (wave * 2 + jj) * 16 + (lane & 15);
            float bb = bias2[cout];
#pragma unroll
            for (int r = 0; r < 8; ++r) {
                float z = acc[mt][jj][r] + bb;
                float sv = z / (1.f + __expf(-z));            // SiLU
                int xg = mt * 16 + mrow + r;
                feat[(((b * 80 + y) * 80 + xg) * 256) + cout] = (_Float16)sv;
            }
        }
}

// ---------------------------------------------------------------------------
// ROI-align 1x1 via TDM gather-mode: 4 tensor_load_to_lds ops gather the 4
// bilinear tap rows (16 samples each, 512B/row) of feat[b] into LDS; combine.
// ---------------------------------------------------------------------------
DEV void samp_geom(float x1, float y1, float rw, float rh, int s, int rows[4], float w[4]) {
    int sy = s >> 2, sx = s & 3;
    float ys = y1 + (sy + 0.5f) * 0.25f * rh;
    float xs = x1 + (sx + 0.5f) * 0.25f * rw;
    bool valid = (ys >= -1.f) && (ys <= 80.f) && (xs >= -1.f) && (xs <= 80.f);
    float yc = fminf(fmaxf(ys, 0.f), 79.f);
    float xc = fminf(fmaxf(xs, 0.f), 79.f);
    float y0f = floorf(yc), x0f = floorf(xc);
    int y0 = (int)y0f, x0 = (int)x0f;
    int y1i = min(y0 + 1, 79), x1i = min(x0 + 1, 79);
    float ly = yc - y0f, lx = xc - x0f, hy = 1.f - ly, hx = 1.f - lx;
    float vf = valid ? 1.f : 0.f;
    rows[0] = y0 * 80 + x0;  rows[1] = y0 * 80 + x1i;
    rows[2] = y1i * 80 + x0; rows[3] = y1i * 80 + x1i;
    w[0] = hy * hx * vf; w[1] = hy * lx * vf; w[2] = ly * hx * vf; w[3] = ly * lx * vf;
}

__launch_bounds__(256)
__global__ void roi_kernel(const _Float16* __restrict__ feat, const float* __restrict__ boxes,
                           float* __restrict__ nodes_f, _Float16* __restrict__ nodes_h) {
    __shared__ _Float16 tile[4][16][256];      // [tap][sample][channel]
    int n = blockIdx.x;                        // 0..511
    int b = n >> 5;
    int c = threadIdx.x;
    const float* bx = boxes + n * 4;
    float x1 = bx[0] * 79.f, y1 = bx[1] * 79.f, x2 = bx[2] * 79.f, y2 = bx[3] * 79.f;
    float rw = fmaxf(x2 - x1, 1.f), rh = fmaxf(y2 - y1, 1.f);

#if __has_builtin(__builtin_amdgcn_tensor_load_to_lds)
    if (threadIdx.x < 32) {                    // TDM ignores EXEC: issue from one wave
        int pk[4][8];
#pragma unroll
        for (int s = 0; s < 16; s += 2) {
            int ra[4], rb[4]; float wa[4], wb[4];
            samp_geom(x1, y1, rw, rh, s, ra, wa);
            samp_geom(x1, y1, rw, rh, s + 1, rb, wb);
#pragma unroll
            for (int t = 0; t < 4; ++t)
                pk[t][s >> 1] = (ra[t] & 0xffff) | (rb[t] << 16);
        }
        typedef __attribute__((address_space(3))) _Float16 lds_f16_t;
        unsigned lds0 = (unsigned)(unsigned long long)(lds_f16_t*)&tile[0][0][0];
        unsigned long long ga = (unsigned long long)(const void*)(feat + (size_t)b * 6400 * 256);
        // D# group1: mask=0, data_size=2B(1), dim0=256, dim1=6400 rows,
        // tile_dim0=256, tile_dim1=16 indices, dim0_stride=256.
        v8i g1 = { 0x00010000, (int)(256u << 16), (int)(6400u << 16),
                   (int)(256u << 16), 16, 256, 0, 0 };
#pragma unroll
        for (int t = 0; t < 4; ++t) {
            // group0: count=1, gather_mode=1 (16-bit idx), lds_addr, global_addr, type=2
            v4u g0 = { 0x80000001u, lds0 + (unsigned)(t * 16 * 256 * 2),
                       (unsigned)ga,
                       (unsigned)((ga >> 32) & 0x01FFFFFFull) | 0x80000000u };
            v4i g2 = { pk[t][0], pk[t][1], pk[t][2], pk[t][3] };
            v4i g3 = { pk[t][4], pk[t][5], pk[t][6], pk[t][7] };
#if defined(__clang_major__) && __clang_major__ >= 23
            __builtin_amdgcn_tensor_load_to_lds(g0, g1, g2, g3, (v8i)0, 0);
#else
            __builtin_amdgcn_tensor_load_to_lds(g0, g1, g2, g3, 0);
#endif
        }
        __builtin_amdgcn_s_wait_tensorcnt(0);
    }
    __syncthreads();
#else
    // Fallback: cooperative direct gather into the same LDS tile.
    for (int s = 0; s < 16; ++s) {
        int rows[4]; float w[4];
        samp_geom(x1, y1, rw, rh, s, rows, w);
#pragma unroll
        for (int t = 0; t < 4; ++t)
            tile[t][s][c] = feat[((size_t)b * 6400 + rows[t]) * 256 + c];
    }
    __syncthreads();
#endif

    float acc = 0.f;
    for (int s = 0; s < 16; ++s) {
        int rows[4]; float w[4];
        samp_geom(x1, y1, rw, rh, s, rows, w);
        acc += w[0] * (float)tile[0][s][c] + w[1] * (float)tile[1][s][c]
             + w[2] * (float)tile[2][s][c] + w[3] * (float)tile[3][s][c];
    }
    acc *= (1.f / 16.f);
    nodes_f[n * 256 + c] = acc;
    nodes_h[n * 256 + c] = (_Float16)acc;
}

// ---------------------------------------------------------------------------
// QKV projections: 3x [512,256]x[256,256] GEMMs with WMMA. Block = 16 rows,
// 8 waves x 6 (matrix, ntile) combos = all 48 output tiles.
// ---------------------------------------------------------------------------
__launch_bounds__(256)
__global__ void qkv_wmma_kernel(const _Float16* __restrict__ nodes_h, const _Float16* __restrict__ lp,
                                const float* __restrict__ bq, const float* __restrict__ bk,
                                const float* __restrict__ bv,
                                _Float16* __restrict__ qb, _Float16* __restrict__ kb,
                                _Float16* __restrict__ vb) {
    __shared__ _Float16 ash[16 * 256];
    int i0 = blockIdx.x * 16;
    int tid = threadIdx.x, lane = tid & 31, wave = tid >> 5;
    for (int i = tid; i < 2048; i += 256)
        ((unsigned int*)ash)[i] = ((const unsigned int*)(nodes_h + i0 * 256))[i];
    __syncthreads();

    v8f acc[6];
#pragma unroll
    for (int t = 0; t < 6; ++t) acc[t] = {};
    int base = (lane & 16) ? 8 : 0;

    for (int ch = 0; ch < 8; ++ch) {
        const _Float16* p = ash + (lane & 15) * 256 + ch * 32 + base;
        v16h af = cat8(*(const v8h*)p, *(const v8h*)(p + 16));
#pragma unroll
        for (int t = 0; t < 6; ++t) {
            int cc = wave * 6 + t;           // 0..47
            int mat = cc >> 4, nt = cc & 15;
            v16h bf = *(const v16h*)(lp + ((((mat * 8 + ch) * 16 + nt) * 32 + lane) << 4));
            acc[t] = __builtin_amdgcn_wmma_f32_16x16x32_f16(false, af, false, bf, (short)0,
                                                            acc[t], false, false);
        }
    }
    int mrow = (lane & 16) ? 8 : 0;
#pragma unroll
    for (int t = 0; t < 6; ++t) {
        int cc = wave * 6 + t;
        int mat = cc >> 4, nt = cc & 15;
        int cout = nt * 16 + (lane & 15);
        const float* bias = (mat == 0) ? bq : (mat == 1) ? bk : bv;
        _Float16* ob = (mat == 0) ? qb : (mat == 1) ? kb : vb;
        float bb = bias[cout];
#pragma unroll
        for (int r = 0; r < 8; ++r)
            ob[(i0 + mrow + r) * 256 + cout] = (_Float16)(acc[t][r] + bb);
    }
}

// ---------------------------------------------------------------------------
// Attention (global over 512 nodes, per head). Flash-style online softmax in
// VALU (only 0.27 GFLOP). Block = (head, 64 rows); K/V head tiles in LDS.
// ---------------------------------------------------------------------------
__launch_bounds__(256)
__global__ void attn_kernel(const _Float16* __restrict__ qb, const _Float16* __restrict__ kb,
                            const _Float16* __restrict__ vb, const float* __restrict__ scale_p,
                            _Float16* __restrict__ msg_h) {
    extern __shared__ _Float16 smem[];
    _Float16* ksh = smem;                       // 512*32 f16
    _Float16* vsh = smem + 512 * 32;            // 512*32 f16
    float* red = (float*)(smem + 2 * 512 * 32); // 64*4*34 f32
    int h = blockIdx.x >> 3, g = blockIdx.x & 7;
    int i0 = g * 64;
    int tid = threadIdx.x;
    float scale = scale_p[0];

    for (int idx = tid; idx < 512 * 32; idx += 256) {
        int j = idx >> 5, d = idx & 31;
        ksh[idx] = kb[j * 256 + h * 32 + d];
        vsh[idx] = vb[j * 256 + h * 32 + d];
    }
    __syncthreads();

    int i = i0 + (tid >> 2);
    int p = tid & 3;
    float q[32], acc[32];
#pragma unroll
    for (int d = 0; d < 32; ++d) { q[d] = (float)qb[i * 256 + h * 32 + d]; acc[d] = 0.f; }
    float mmax = -3.0e38f, ssum = 0.f;

    for (int j = p * 128; j < p * 128 + 128; ++j) {
        float s = 0.f;
#pragma unroll
        for (int d = 0; d < 32; ++d) s += q[d] * (float)ksh[j * 32 + d];
        s *= scale;
        float mn = fmaxf(mmax, s);
        float corr = __expf(mmax - mn);
        float e = __expf(s - mn);
        ssum = ssum * corr + e;
#pragma unroll
        for (int d = 0; d < 32; ++d) acc[d] = acc[d] * corr + e * (float)vsh[j * 32 + d];
        mmax = mn;
    }
    float* rp = red + ((tid >> 2) * 4 + p) * 34;
    rp[0] = mmax; rp[1] = ssum;
#pragma unroll
    for (int d = 0; d < 32; ++d) rp[2 + d] = acc[d];
    __syncthreads();

    if (p == 0) {
        float* rr = red + (tid >> 2) * 4 * 34;
        float M = -3.0e38f;
#pragma unroll
        for (int t = 0; t < 4; ++t) M = fmaxf(M, rr[t * 34]);
        float S = 0.f, out[32];
#pragma unroll
        for (int d = 0; d < 32; ++d) out[d] = 0.f;
#pragma unroll
        for (int t = 0; t < 4; ++t) {
            float cc = __expf(rr[t * 34] - M);
            S += rr[t * 34 + 1] * cc;
#pragma unroll
            for (int d = 0; d < 32; ++d) out[d] += rr[t * 34 + 2 + d] * cc;
        }
        float inv = 1.f / S;
#pragma unroll
        for (int d = 0; d < 32; ++d)
            msg_h[i * 256 + h * 32 + d] = (_Float16)(out[d] * inv);
    }
}

// ---------------------------------------------------------------------------
// Out-projection (WMMA) + residual + LayerNorm -> enh f32 [512][256].
// ---------------------------------------------------------------------------
__launch_bounds__(256)
__global__ void outproj_ln_kernel(const _Float16* __restrict__ msg_h, const _Float16* __restrict__ wo_p,
                                  const float* __restrict__ nodes_f, const float* __restrict__ bo,
                                  const float* __restrict__ ln_g, const float* __restrict__ ln_b,
                                  float* __restrict__ enh) {
    __shared__ _Float16 ash[16 * 256];
    __shared__ float ysh[16 * 256];
    int i0 = blockIdx.x * 16;
    int tid = threadIdx.x, lane = tid & 31, wave = tid >> 5;
    for (int i = tid; i < 2048; i += 256)
        ((unsigned int*)ash)[i] = ((const unsigned int*)(msg_h + i0 * 256))[i];
    __syncthreads();

    v8f acc[2]; acc[0] = {}; acc[1] = {};
    int base = (lane & 16) ? 8 : 0;
    for (int ch = 0; ch < 8; ++ch) {
        const _Float16* p = ash + (lane & 15) * 256 + ch * 32 + base;
        v16h af = cat8(*(const v8h*)p, *(const v8h*)(p + 16));
#pragma unroll
        for (int jj = 0; jj < 2; ++jj) {
            int nt = wave * 2 + jj;
            v16h bf = *(const v16h*)(wo_p + (((ch * 16 + nt) * 32 + lane) << 4));
            acc[jj] = __builtin_amdgcn_wmma_f32_16x16x32_f16(false, af, false, bf, (short)0,
                                                             acc[jj], false, false);
        }
    }
    int mrow = (lane & 16) ? 8 : 0;
#pragma unroll
    for (int jj = 0; jj < 2; ++jj) {
        int cout = (wave * 2 + jj) * 16 + (lane & 15);
#pragma unroll
        for (int r = 0; r < 8; ++r) {
            int m = mrow + r;
            ysh[m * 256 + cout] = acc[jj][r] + nodes_f[(i0 + m) * 256 + cout] + bo[cout];
        }
    }
    __syncthreads();
    if (tid < 16) {
        float mu = 0.f;
        for (int c = 0; c < 256; ++c) mu += ysh[tid * 256 + c];
        mu *= (1.f / 256.f);
        float var = 0.f;
        for (int c = 0; c < 256; ++c) { float d = ysh[tid * 256 + c] - mu; var += d * d; }
        var *= (1.f / 256.f);
        float inv = rsqrtf(var + 1e-5f);
        for (int c = 0; c < 256; ++c)
            enh[(i0 + tid) * 256 + c] = (ysh[tid * 256 + c] - mu) * inv * ln_g[c] + ln_b[c];
    }
}

// ---------------------------------------------------------------------------
// Pooling + rec 1x1 conv + BN2 -> per-sample bias ybias[16][256]. One block.
// ---------------------------------------------------------------------------
__launch_bounds__(256)
__global__ void pool_kernel(const float* __restrict__ enh, const float* __restrict__ rec_w,
                            const float* __restrict__ rec_b, const float* __restrict__ g2,
                            const float* __restrict__ b2, const float* __restrict__ m2,
                            const float* __restrict__ v2, float* __restrict__ ybias) {
    __shared__ float rm[512];
    __shared__ float wts[512];
    __shared__ float wsum[4096];
    int tid = threadIdx.x;
    for (int r = tid; r < 512; r += 256) {
        float s = 0.f;
        for (int c = 0; c < 256; ++c) s += enh[r * 256 + c];
        rm[r] = s * (1.f / 256.f);
    }
    __syncthreads();
    if (tid < 16) {
        float M = -3.0e38f;
        for (int k = 0; k < 32; ++k) M = fmaxf(M, rm[tid * 32 + k]);
        float S = 0.f;
        for (int k = 0; k < 32; ++k) { float e = __expf(rm[tid * 32 + k] - M); wts[tid * 32 + k] = e; S += e; }
        float inv = 1.f / S;
        for (int k = 0; k < 32; ++k) wts[tid * 32 + k] *= inv;
    }
    __syncthreads();
    for (int idx = tid; idx < 4096; idx += 256) {
        int b = idx >> 8, c = idx & 255;
        float s = 0.f;
        for (int k = 0; k < 32; ++k) s += enh[((b * 32 + k) * 256) + c] * wts[b * 32 + k];
        wsum[idx] = s;
    }
    __syncthreads();
    for (int idx = tid; idx < 4096; idx += 256) {
        int b = idx >> 8, c = idx & 255;
        float s = 0.f;
        for (int cp = 0; cp < 256; ++cp) s += wsum[b * 256 + cp] * rec_w[c * 256 + cp];
        s += rec_b[c];
        float sc = g2[c] * rsqrtf(v2[c] + 1e-5f);
        ybias[idx] = s * sc + (b2[c] - m2[c] * sc);
    }
}

// ---------------------------------------------------------------------------
// Final broadcast residual: out = x + ybias[b][c], float4 streaming.
// ---------------------------------------------------------------------------
__launch_bounds__(256)
__global__ void add_kernel(const float* __restrict__ x, const float* __restrict__ ybias,
                           float* __restrict__ out) {
    int idx = blockIdx.x * 256 + threadIdx.x;     // float4 index
    if (idx >= 26214400 / 4) return;
    __builtin_prefetch(((const char*)x) + (size_t)idx * 16 + 8192, 0, 0);
    int bc = (idx * 4) / 6400;                    // (b*256 + c); 6400 % 4 == 0
    float yb = ybias[bc];
    float4 xv = ((const float4*)x)[idx];
    float4 o = make_float4(xv.x + yb, xv.y + yb, xv.z + yb, xv.w + yb);
    ((float4*)out)[idx] = o;
}

// ---------------------------------------------------------------------------
extern "C" void kernel_launch(void* const* d_in, const int* in_sizes, int n_in,
                              void* d_out, int out_size, void* d_ws, size_t ws_size,
                              hipStream_t stream) {
    const float* x      = (const float*)d_in[0];
    const float* boxes  = (const float*)d_in[1];
    const float* conv_w = (const float*)d_in[2];
    const float* conv_b = (const float*)d_in[3];
    const float* bn1_g  = (const float*)d_in[4];
    const float* bn1_b  = (const float*)d_in[5];
    const float* bn1_m  = (const float*)d_in[6];
    const float* bn1_v  = (const float*)d_in[7];
    const float* wq     = (const float*)d_in[8];
    const float* bq     = (const float*)d_in[9];
    const float* wk     = (const float*)d_in[10];
    const float* bkp    = (const float*)d_in[11];
    const float* wv     = (const float*)d_in[12];
    const float* bv     = (const float*)d_in[13];
    const float* scale  = (const float*)d_in[14];
    const float* wo     = (const float*)d_in[15];
    const float* bo     = (const float*)d_in[16];
    const float* ln_g   = (const float*)d_in[17];
    const float* ln_b   = (const float*)d_in[18];
    const float* rec_w  = (const float*)d_in[19];
    const float* rec_b  = (const float*)d_in[20];
    const float* bn2_g  = (const float*)d_in[21];
    const float* bn2_b  = (const float*)d_in[22];
    const float* bn2_m  = (const float*)d_in[23];
    const float* bn2_v  = (const float*)d_in[24];
    float* out = (float*)d_out;

    char* ws = (char*)d_ws;
    size_t off = 0;
    auto alloc = [&](size_t bytes) -> void* {
        void* p = ws + off;
        off += (bytes + 255) & ~(size_t)255;
        return p;
    };
    _Float16* feat    = (_Float16*)alloc((size_t)16 * 80 * 80 * 256 * 2);  // 52.4 MB
    _Float16* convWp  = (_Float16*)alloc((size_t)9 * 8 * 16 * 32 * 16 * 2);
    float*    bias2   = (float*)alloc(256 * 4);
    _Float16* linWp   = (_Float16*)alloc((size_t)4 * 8 * 16 * 32 * 16 * 2);
    float*    nodes_f = (float*)alloc(512 * 256 * 4);
    _Float16* nodes_h = (_Float16*)alloc(512 * 256 * 2);
    _Float16* qb      = (_Float16*)alloc(512 * 256 * 2);
    _Float16* kb      = (_Float16*)alloc(512 * 256 * 2);
    _Float16* vb      = (_Float16*)alloc(512 * 256 * 2);
    _Float16* msg_h   = (_Float16*)alloc(512 * 256 * 2);
    float*    enh     = (float*)alloc(512 * 256 * 4);
    float*    ybias   = (float*)alloc(16 * 256 * 4);

    pack_conv_kernel<<<(9 * 8 * 16 * 32 * 16 + 255) / 256, 256, 0, stream>>>(
        conv_w, conv_b, bn1_g, bn1_b, bn1_m, bn1_v, convWp, bias2);
    pack_lin_kernel<<<(4 * 8 * 16 * 32 * 16 + 255) / 256, 256, 0, stream>>>(wq, wk, wv, wo, linWp);

    conv_wmma_kernel<<<16 * 80, 256, 0, stream>>>(x, convWp, bias2, feat);
    roi_kernel<<<512, 256, 0, stream>>>(feat, boxes, nodes_f, nodes_h);
    qkv_wmma_kernel<<<512 / 16, 256, 0, stream>>>(nodes_h, linWp, bq, bkp, bv, qb, kb, vb);

    size_t attn_smem = 2 * 512 * 32 * sizeof(_Float16) + 64 * 4 * 34 * sizeof(float);
    attn_kernel<<<8 * 8, 256, attn_smem, stream>>>(qb, kb, vb, scale, msg_h);

    const _Float16* wo_p = linWp + (size_t)3 * 8 * 16 * 32 * 16;
    outproj_ln_kernel<<<512 / 16, 256, 0, stream>>>(msg_h, wo_p, nodes_f, bo, ln_g, ln_b, enh);
    pool_kernel<<<1, 256, 0, stream>>>(enh, rec_w, rec_b, bn2_g, bn2_b, bn2_m, bn2_v, ybias);
    add_kernel<<<(26214400 / 4 + 255) / 256, 256, 0, stream>>>(x, ybias, out);

    (void)in_sizes; (void)n_in; (void)out_size; (void)ws_size;
}
